// SVDHead_20229295964467
// MI455X (gfx1250) — matrix operand order repaired and checked by reading, not compiled, not run on hardware
//
#include <hip/hip_runtime.h>
#include <hip/hip_bf16.h>

// ---------------------------------------------------------------------------
// SVDHead (DCP-style): fused attention-correspondence + batched 3x3 Kabsch.
// B=16, D=512, N=1024.
//
// Kernel 0: one-time transpose + bf16 hi/lo split of both embeddings into
//           workspace, K-major ([B,N,D] bf16). Removes ALL conversion work
//           from the GEMM hot loop (previously re-done 64x per batch).
// Kernel 1: per (batch, 16-row n-tile) workgroup, flash-style fused
//           logits-GEMM (bf16 hi/lo split WMMA) + online softmax + src_corr.
//           A-tile staged to LDS via global_load_async_to_lds_b128 (pure DMA),
//           B fragments streamed directly from L2 as global_load_b128.
// Kernel 2: per-batch covariance + Jacobi 3x3 SVD -> R, t.
//
// Workspace: 192KB corr + 64MB bf16 embeddings (hi/lo x src/tgt).
// ---------------------------------------------------------------------------

typedef __attribute__((ext_vector_type(16))) __bf16 v16bf;
typedef __attribute__((ext_vector_type(8)))  float  v8f;

constexpr int B_ = 16;
constexpr int D_ = 512;
constexpr int N_ = 1024;

constexpr int TN    = 16;               // n-rows per workgroup
constexpr int WAVES = 4;                // waves per workgroup (wave32)
constexpr int MSEG  = N_ / WAVES;       // 256 columns per wave

// Kernel-1 LDS layout (bytes). A rows padded to 1040B -> lane r's fragment
// base lands at 16r mod 256: all 16 lanes hit distinct 16B bank slots.
constexpr int ASTRB = D_ * 2 + 16;                  // 1040 B per A row (bf16)
constexpr int AHI_B = 0;
constexpr int ALO_B = AHI_B + TN * ASTRB;           // 16640
constexpr int RED_B = ALO_B + TN * ASTRB;           // 33280
constexpr size_t LDS_BYTES = RED_B + WAVES * TN * 8 * sizeof(float); // 35328

// Workspace layout (bytes)
constexpr size_t CORR_BYTES = (size_t)B_ * 3 * N_ * sizeof(float);   // 196608
constexpr size_t CONV_ELEMS = (size_t)B_ * N_ * D_;                  // 8M bf16

// hi = truncated-bf16(f), lo = bf16(f - hi); returned as u16 bit patterns.
__device__ inline void split_bits(float f, unsigned& hb, unsigned& lb) {
  const unsigned u = __builtin_bit_cast(unsigned, f);
  hb = u >> 16;
  const float hf = __builtin_bit_cast(float, hb << 16);
  lb = __builtin_bit_cast(unsigned, f - hf) >> 16;
}

// Build a v16bf fragment from two 16B chunks `gap` bytes apart (LDS or global).
union FragU { uint4 q[2]; v16bf v; };
__device__ inline v16bf ld_frag2(const char* p, int gap) {
  FragU f;
  f.q[0] = *(const uint4*)(p);
  f.q[1] = *(const uint4*)(p + gap);
  return f.v;
}

// ---------------------------------------------------------------------------
// Kernel 0: [B,D,N] fp32 -> [B,N,D] bf16 hi / lo  (64x64 LDS-tiled transpose)
// ---------------------------------------------------------------------------
__global__ void __launch_bounds__(256)
convert_transpose_kernel(const float* __restrict__ src_emb,
                         const float* __restrict__ tgt_emb,
                         unsigned short* __restrict__ srcT_hi,
                         unsigned short* __restrict__ srcT_lo,
                         unsigned short* __restrict__ tgtT_hi,
                         unsigned short* __restrict__ tgtT_lo)
{
  __shared__ float tile[64][68];
  const int n0  = blockIdx.x * 64;
  const int d0  = blockIdx.y * 64;
  const int b   = blockIdx.z & 15;
  const int arr = blockIdx.z >> 4;
  const float*    in = (arr ? tgt_emb : src_emb) + (size_t)b * D_ * N_;
  unsigned short* oh = (arr ? tgtT_hi : srcT_hi) + (size_t)b * N_ * D_;
  unsigned short* ol = (arr ? tgtT_lo : srcT_lo) + (size_t)b * N_ * D_;
  const int tid = threadIdx.x;

  { // coalesced fp32 tile load (rows of n)
    const int r4 = tid >> 4;            // 0..15
    const int c4 = (tid & 15) * 4;
#pragma unroll
    for (int p = 0; p < 4; ++p) {
      const int row = p * 16 + r4;
      const float4 v = *(const float4*)(in + (size_t)(d0 + row) * N_ + n0 + c4);
      tile[row][c4 + 0] = v.x; tile[row][c4 + 1] = v.y;
      tile[row][c4 + 2] = v.z; tile[row][c4 + 3] = v.w;
    }
  }
  __syncthreads();

  { // transposed + split store: out[n][d], d-pairs packed as u32
    const int nl = tid & 63;
    const int dp = tid >> 6;            // 0..3
    char* rh = (char*)(oh + (size_t)(n0 + nl) * D_ + d0);
    char* rl = (char*)(ol + (size_t)(n0 + nl) * D_ + d0);
#pragma unroll
    for (int q = 0; q < 8; ++q) {
      const int dl = 2 * (q * 4 + dp);  // even, 0..62
      unsigned h0, l0, h1, l1;
      split_bits(tile[dl][nl],     h0, l0);
      split_bits(tile[dl + 1][nl], h1, l1);
      *(unsigned*)(rh + dl * 2) = h0 | (h1 << 16);
      *(unsigned*)(rl + dl * 2) = l0 | (l1 << 16);
    }
  }
}

// ---------------------------------------------------------------------------
// Kernel 1: fused logits-GEMM + online softmax + correspondence
// ---------------------------------------------------------------------------
__global__ void __launch_bounds__(128)
fused_attn_corr_kernel(const unsigned short* __restrict__ srcT_hi, // [B,N,D] bf16
                       const unsigned short* __restrict__ srcT_lo,
                       const unsigned short* __restrict__ tgtT_hi, // [B,N,D] bf16
                       const unsigned short* __restrict__ tgtT_lo,
                       const float* __restrict__ tgt,              // [B,3,N]
                       const float* __restrict__ temperature,      // [B]
                       float* __restrict__ corr_out)               // [B,3,N]
{
  extern __shared__ char lds8[];
  float* red = (float*)(lds8 + RED_B);  // [WAVES][TN][8]

  const int n0   = blockIdx.x * TN;
  const int b    = blockIdx.y;
  const int tid  = threadIdx.x;
  const int wave = tid >> 5;
  const int lane = tid & 31;
  const int half = (lane >> 4) & 1;     // which 16-lane half
  const int r    = lane & 15;           // row (A) / col (B/C) within tile

  const float scale = temperature[b] * 0.04419417382415922f; // 1/sqrt(512)

  // ==== Stage A-tile hi/lo into LDS: pure async DMA (16 rows x 1KB each) ====
  {
    const char* gh = (const char*)srcT_hi + ((size_t)b * N_ + n0) * (D_ * 2);
    const char* gl = (const char*)srcT_lo + ((size_t)b * N_ + n0) * (D_ * 2);
#pragma unroll
    for (int i = 0; i < 8; ++i) {
      const int idx  = tid + i * 128;   // 0..1023
      const int nrow = idx >> 6;        // 0..15
      const int seg  = idx & 63;        // 16B segment within 1KB row
      const unsigned long long gah =
          (unsigned long long)(const void*)(gh + (size_t)nrow * (D_ * 2) + seg * 16);
      const unsigned long long gal =
          (unsigned long long)(const void*)(gl + (size_t)nrow * (D_ * 2) + seg * 16);
      const unsigned lah =
          (unsigned)(size_t)(const void*)(lds8 + AHI_B + nrow * ASTRB + seg * 16);
      const unsigned lal =
          (unsigned)(size_t)(const void*)(lds8 + ALO_B + nrow * ASTRB + seg * 16);
      asm volatile("global_load_async_to_lds_b128 %0, %1, off"
                   :: "v"(lah), "v"(gah) : "memory");
      asm volatile("global_load_async_to_lds_b128 %0, %1, off"
                   :: "v"(lal), "v"(gal) : "memory");
    }
  }
  asm volatile("s_wait_asynccnt 0x0" ::: "memory");
  __syncthreads();

  // ==== Online-softmax running state (per lane; rows v / v+8 per half) ======
  float rmax[8], rsum[8], c0a[8], c1a[8], c2a[8];
#pragma unroll
  for (int v = 0; v < 8; ++v) { rmax[v] = -1e30f; rsum[v] = 0.f; c0a[v] = c1a[v] = c2a[v] = 0.f; }

  const char* BgHi = (const char*)tgtT_hi + (size_t)b * N_ * (D_ * 2);
  const char* BgLo = (const char*)tgtT_lo + (size_t)b * N_ * (D_ * 2);
  const char* aHip = lds8 + AHI_B + r * ASTRB + half * 16; // + kk*2 per step
  const char* aLop = lds8 + ALO_B + r * ASTRB + half * 16;

  for (int t = 0; t < MSEG / 16; ++t) {          // 16 m-tiles per wave
    const int mb = wave * MSEG + t * 16;
    const char* bRowH = BgHi + (size_t)(mb + r) * (D_ * 2) + half * 16;
    const char* bRowL = BgLo + (size_t)(mb + r) * (D_ * 2) + half * 16;

    if (t + 1 < MSEG / 16) {   // prefetch next m-tile's B rows
      __builtin_prefetch(bRowH + (size_t)16 * D_ * 2, 0, 1);
      __builtin_prefetch(bRowL + (size_t)16 * D_ * 2, 0, 1);
    }

    v8f acc;
#pragma unroll
    for (int v = 0; v < 8; ++v) acc[v] = 0.f;

    // A frag: lanes 0-15 want K {kk..kk+7 , kk+16..kk+23} (half*16B shift for
    // lanes 16-31): two b128 loads 32B apart. B frag: same K order for lane r's
    // column: two b128 loads 32B apart, straight from global bf16.
#pragma unroll 4
    for (int kk = 0; kk < D_; kk += 32) {
      const v16bf ahi = ld_frag2(aHip + kk * 2, 32);
      const v16bf alo = ld_frag2(aLop + kk * 2, 32);
      const v16bf bhi = ld_frag2(bRowH + kk * 2, 32);
      const v16bf blo = ld_frag2(bRowL + kk * 2, 32);

      acc = __builtin_amdgcn_wmma_f32_16x16x32_bf16(false, ahi, false, bhi, (short)0, acc, false, false);
      acc = __builtin_amdgcn_wmma_f32_16x16x32_bf16(false, ahi, false, blo, (short)0, acc, false, false);
      acc = __builtin_amdgcn_wmma_f32_16x16x32_bf16(false, alo, false, bhi, (short)0, acc, false, false);
    }

    // -- online softmax + tgt-weighted accumulation for this 16x16 tile ------
    const int m = mb + r;
    const float tx = tgt[(size_t)b * 3 * N_ + 0 * N_ + m];
    const float ty = tgt[(size_t)b * 3 * N_ + 1 * N_ + m];
    const float tz = tgt[(size_t)b * 3 * N_ + 2 * N_ + m];

#pragma unroll
    for (int v = 0; v < 8; ++v) {
      float x = acc[v] * scale;
      float tmax = x;
      tmax = fmaxf(tmax, __shfl_xor(tmax, 1, 32));
      tmax = fmaxf(tmax, __shfl_xor(tmax, 2, 32));
      tmax = fmaxf(tmax, __shfl_xor(tmax, 4, 32));
      tmax = fmaxf(tmax, __shfl_xor(tmax, 8, 32));
      const float nm = fmaxf(rmax[v], tmax);
      const float f  = __expf(rmax[v] - nm);
      const float p  = __expf(x - nm);
      rmax[v] = nm;
      rsum[v] = rsum[v] * f + p;
      c0a[v]  = c0a[v]  * f + p * tx;
      c1a[v]  = c1a[v]  * f + p * ty;
      c2a[v]  = c2a[v]  * f + p * tz;
    }
  }

  // ---- reduce sums across the 16 lanes holding each row ---------------------
#pragma unroll
  for (int v = 0; v < 8; ++v) {
#pragma unroll
    for (int msk = 1; msk <= 8; msk <<= 1) {
      rsum[v] += __shfl_xor(rsum[v], msk, 32);
      c0a[v]  += __shfl_xor(c0a[v],  msk, 32);
      c1a[v]  += __shfl_xor(c1a[v],  msk, 32);
      c2a[v]  += __shfl_xor(c2a[v],  msk, 32);
    }
  }
  if (r == 0) {
#pragma unroll
    for (int v = 0; v < 8; ++v) {
      const int row = v + half * 8;
      float* e = red + (wave * TN + row) * 8;
      e[0] = rmax[v]; e[1] = rsum[v]; e[2] = c0a[v]; e[3] = c1a[v]; e[4] = c2a[v];
    }
  }
  __syncthreads();

  // ---- merge the 4 wave partials (softmax merge) and emit src_corr ---------
  if (tid < TN) {
    const int row = tid;
    float M = -1e30f;
    for (int w = 0; w < WAVES; ++w) M = fmaxf(M, red[(w * TN + row) * 8 + 0]);
    float S = 0.f, C0 = 0.f, C1 = 0.f, C2 = 0.f;
    for (int w = 0; w < WAVES; ++w) {
      const float* e = red + (w * TN + row) * 8;
      const float f = __expf(e[0] - M);
      S += e[1] * f; C0 += e[2] * f; C1 += e[3] * f; C2 += e[4] * f;
    }
    const float inv = 1.0f / S;
    const size_t o = (size_t)b * 3 * N_ + (n0 + row);
    corr_out[o + 0 * N_] = C0 * inv;
    corr_out[o + 1 * N_] = C1 * inv;
    corr_out[o + 2 * N_] = C2 * inv;
  }
}

// ---------------------------------------------------------------------------
// Kernel 2: per-batch covariance + 3x3 Kabsch (Jacobi eigen of H^T H).
// ---------------------------------------------------------------------------
__device__ inline float det3(const float M[3][3]) {
  return M[0][0] * (M[1][1] * M[2][2] - M[1][2] * M[2][1])
       - M[0][1] * (M[1][0] * M[2][2] - M[1][2] * M[2][0])
       + M[0][2] * (M[1][0] * M[2][1] - M[1][1] * M[2][0]);
}

__global__ void __launch_bounds__(64)
kabsch_svd_kernel(const float* __restrict__ src,   // [B,3,N]
                  const float* __restrict__ corr,  // [B,3,N]
                  float* __restrict__ out)         // R[B,3,3] ++ t[B,3]
{
  __shared__ float part[15][64];
  const int b   = blockIdx.x;
  const int tid = threadIdx.x;
  const float* S = src  + (size_t)b * 3 * N_;
  const float* C = corr + (size_t)b * 3 * N_;

  float s0 = 0, s1 = 0, s2 = 0, q0 = 0, q1 = 0, q2 = 0;
  float hp[9] = {0, 0, 0, 0, 0, 0, 0, 0, 0};
  for (int n = tid; n < N_; n += 64) {
    const float sx = S[n], sy = S[N_ + n], sz = S[2 * N_ + n];
    const float cx = C[n], cy = C[N_ + n], cz = C[2 * N_ + n];
    s0 += sx; s1 += sy; s2 += sz; q0 += cx; q1 += cy; q2 += cz;
    hp[0] += sx * cx; hp[1] += sx * cy; hp[2] += sx * cz;
    hp[3] += sy * cx; hp[4] += sy * cy; hp[5] += sy * cz;
    hp[6] += sz * cx; hp[7] += sz * cy; hp[8] += sz * cz;
  }
  part[0][tid] = s0; part[1][tid] = s1; part[2][tid] = s2;
  part[3][tid] = q0; part[4][tid] = q1; part[5][tid] = q2;
#pragma unroll
  for (int k = 0; k < 9; ++k) part[6 + k][tid] = hp[k];
  __syncthreads();

  if (tid != 0) return;

  float tot[15];
  for (int k = 0; k < 15; ++k) {
    float a = 0.f;
    for (int i = 0; i < 64; ++i) a += part[k][i];
    tot[k] = a;
  }
  const float invN = 1.0f / (float)N_;
  float ms[3] = {tot[0] * invN, tot[1] * invN, tot[2] * invN};
  float mc[3] = {tot[3] * invN, tot[4] * invN, tot[5] * invN};
  float H[3][3];
  {
    int k = 6;
    for (int i = 0; i < 3; ++i)
      for (int j = 0; j < 3; ++j, ++k)
        H[i][j] = tot[k] - (float)N_ * ms[i] * mc[j];
  }

  // Symmetric S = H^T H ; eigenvectors V via cyclic Jacobi.
  float Sm[3][3];
  for (int i = 0; i < 3; ++i)
    for (int j = 0; j < 3; ++j)
      Sm[i][j] = H[0][i] * H[0][j] + H[1][i] * H[1][j] + H[2][i] * H[2][j];
  float V[3][3] = {{1, 0, 0}, {0, 1, 0}, {0, 0, 1}};
  const int PP[3] = {0, 0, 1}, QQ[3] = {1, 2, 2};
  for (int sweep = 0; sweep < 10; ++sweep) {
    for (int pr = 0; pr < 3; ++pr) {
      const int p = PP[pr], q = QQ[pr];
      const float apq = Sm[p][q];
      if (fabsf(apq) < 1e-20f) continue;
      const float tau = (Sm[q][q] - Sm[p][p]) / (2.0f * apq);
      const float tt  = (tau >= 0.f ? 1.f : -1.f) / (fabsf(tau) + sqrtf(1.f + tau * tau));
      const float cc  = rsqrtf(1.f + tt * tt);
      const float ss  = tt * cc;
      for (int k = 0; k < 3; ++k) {   // right rotation of columns p,q
        const float a = Sm[k][p], c2 = Sm[k][q];
        Sm[k][p] = cc * a - ss * c2;
        Sm[k][q] = ss * a + cc * c2;
      }
      for (int k = 0; k < 3; ++k) {   // left rotation of rows p,q
        const float a = Sm[p][k], c2 = Sm[q][k];
        Sm[p][k] = cc * a - ss * c2;
        Sm[q][k] = ss * a + cc * c2;
      }
      for (int k = 0; k < 3; ++k) {   // accumulate V
        const float a = V[k][p], c2 = V[k][q];
        V[k][p] = cc * a - ss * c2;
        V[k][q] = ss * a + cc * c2;
      }
    }
  }

  // Sort eigenpairs descending -> Vs
  float e[3] = {Sm[0][0], Sm[1][1], Sm[2][2]};
  int idx[3] = {0, 1, 2};
  for (int i = 0; i < 2; ++i)
    for (int j = i + 1; j < 3; ++j)
      if (e[idx[j]] > e[idx[i]]) { int tswp = idx[i]; idx[i] = idx[j]; idx[j] = tswp; }
  float Vs[3][3];
  for (int rr = 0; rr < 3; ++rr)
    for (int c = 0; c < 3; ++c)
      Vs[rr][c] = V[rr][idx[c]];

  // U columns = normalize(H * v_c) with degenerate fallbacks.
  float U[3][3];
  for (int c = 0; c < 3; ++c) {
    float uv[3];
    for (int rr = 0; rr < 3; ++rr)
      uv[rr] = H[rr][0] * Vs[0][c] + H[rr][1] * Vs[1][c] + H[rr][2] * Vs[2][c];
    const float nn = sqrtf(uv[0] * uv[0] + uv[1] * uv[1] + uv[2] * uv[2]);
    if (nn > 1e-12f) {
      const float inv = 1.0f / nn;
      for (int rr = 0; rr < 3; ++rr) U[rr][c] = uv[rr] * inv;
    } else if (c == 0) {
      U[0][0] = 1.f; U[1][0] = 0.f; U[2][0] = 0.f;
    } else if (c == 1) {
      const float ax = (fabsf(U[0][0]) < 0.9f) ? 1.f : 0.f;
      const float ay = 1.f - ax;
      float wx = U[1][0] * 0.f - U[2][0] * ay;
      float wy = U[2][0] * ax - U[0][0] * 0.f;
      float wz = U[0][0] * ay - U[1][0] * ax;
      const float wn = rsqrtf(fmaxf(wx * wx + wy * wy + wz * wz, 1e-20f));
      U[0][1] = wx * wn; U[1][1] = wy * wn; U[2][1] = wz * wn;
    } else {
      U[0][2] = U[1][0] * U[2][1] - U[2][0] * U[1][1];
      U[1][2] = U[2][0] * U[0][1] - U[0][0] * U[2][1];
      U[2][2] = U[0][0] * U[1][1] - U[1][0] * U[0][1];
    }
  }

  const float d = (det3(Vs) * det3(U)) >= 0.f ? 1.f : -1.f;
  const float dd[3] = {1.f, 1.f, d};

  float R[3][3];
  for (int i = 0; i < 3; ++i)
    for (int j = 0; j < 3; ++j) {
      float a = 0.f;
      for (int k = 0; k < 3; ++k) a += Vs[i][k] * dd[k] * U[j][k];
      R[i][j] = a;
    }

  for (int i = 0; i < 3; ++i)
    for (int j = 0; j < 3; ++j)
      out[b * 9 + i * 3 + j] = R[i][j];
  for (int i = 0; i < 3; ++i) {
    const float ti = -(R[i][0] * ms[0] + R[i][1] * ms[1] + R[i][2] * ms[2]) + mc[i];
    out[B_ * 9 + b * 3 + i] = ti;
  }
}

// ---------------------------------------------------------------------------
extern "C" void kernel_launch(void* const* d_in, const int* in_sizes, int n_in,
                              void* d_out, int out_size, void* d_ws, size_t ws_size,
                              hipStream_t stream) {
  (void)in_sizes; (void)n_in; (void)out_size; (void)ws_size;
  const float* src_emb = (const float*)d_in[0];   // [16,512,1024]
  const float* tgt_emb = (const float*)d_in[1];   // [16,512,1024]
  const float* src     = (const float*)d_in[2];   // [16,3,1024]
  const float* tgt     = (const float*)d_in[3];   // [16,3,1024]
  const float* temp    = (const float*)d_in[4];   // [16]
  float* out = (float*)d_out;                     // R (144) ++ t (48)

  char* ws = (char*)d_ws;
  float*          corr    = (float*)ws;                                  // 192KB
  unsigned short* srcT_hi = (unsigned short*)(ws + CORR_BYTES);          // 16MB each
  unsigned short* srcT_lo = srcT_hi + CONV_ELEMS;
  unsigned short* tgtT_hi = srcT_lo + CONV_ELEMS;
  unsigned short* tgtT_lo = tgtT_hi + CONV_ELEMS;

  convert_transpose_kernel<<<dim3(N_ / 64, D_ / 64, B_ * 2), dim3(256), 0, stream>>>(
      src_emb, tgt_emb, srcT_hi, srcT_lo, tgtT_hi, tgtT_lo);

  fused_attn_corr_kernel<<<dim3(N_ / TN, B_), dim3(128), LDS_BYTES, stream>>>(
      srcT_hi, srcT_lo, tgtT_hi, tgtT_lo, tgt, temp, corr);

  kabsch_svd_kernel<<<dim3(B_), dim3(64), 0, stream>>>(src, corr, out);
}
